// PixachuSelfAttention_38465727103577
// MI455X (gfx1250) — compile-verified, matrix operand order checked by reading, not run on hardware
//
#include <hip/hip_runtime.h>
#include <hip/hip_bf16.h>

// ---------------- constants ----------------
#define H_DIMC   1152
#define N_HEADSC 12
#define HEAD_DIMC 96
#define SEQC     1025
#define BATCHC   8
#define TOKC     (BATCHC * SEQC)     // 8200 real token rows
#define MT32     257                 // ceil(8200/32) -> 8224 rows computed
#define MPADC    8448                // padded token rows in workspace buffers
#define SPADC    1056                // padded key count (33 * 32)
#define KTILES   33                  // key tiles of 32
#define QT32     33                  // query tiles of 32 (33*32 = 1056)
#define N4TILES  18                  // 1152 / 64

// WMMA kernels: 4 waves/block (1 per SIMD32) and explicit permission to use the
// full 1024-VGPR file -> no spills; parallelism comes from thousands of blocks.
#define MMA_BOUNDS __launch_bounds__(128) __attribute__((amdgpu_waves_per_eu(1)))

typedef __bf16     bf16_t;
typedef _Float16   f16_t;
typedef __attribute__((ext_vector_type(16))) __bf16   v16bf;
typedef __attribute__((ext_vector_type(8)))  __bf16   v8bf;
typedef __attribute__((ext_vector_type(16))) _Float16 v16h;
typedef __attribute__((ext_vector_type(8)))  _Float16 v8h;
typedef __attribute__((ext_vector_type(8)))  float    v8f;

// Build a 16-element WMMA operand from two contiguous 8-element 16B chunks.
// Element i of the operand corresponds to K = kbase + (i%8) + 16*(i/8),
// matching the CDNA5 16-bit A/B VGPR striping (lane half selects kbase).
__device__ __forceinline__ v16bf ld_pair_bf(const bf16_t* lo, const bf16_t* hi) {
    v8bf a = *(const v8bf*)lo;
    v8bf b = *(const v8bf*)hi;
    return __builtin_shufflevector(a, b, 0,1,2,3,4,5,6,7,8,9,10,11,12,13,14,15);
}
__device__ __forceinline__ v16h ld_pair_h(const f16_t* lo, const f16_t* hi) {
    v8h a = *(const v8h*)lo;
    v8h b = *(const v8h*)hi;
    return __builtin_shufflevector(a, b, 0,1,2,3,4,5,6,7,8,9,10,11,12,13,14,15);
}

// ---------------- prep kernels ----------------

__global__ void prep_x_kernel(const float* __restrict__ X, bf16_t* __restrict__ Xb) {
    long long idx = (long long)blockIdx.x * blockDim.x + threadIdx.x;
    if (idx >= (long long)MPADC * H_DIMC) return;
    int row = (int)(idx / H_DIMC);
    Xb[idx] = (row < TOKC) ? (bf16_t)X[idx] : (bf16_t)0.0f;
}

__global__ void prep_w_kernel(const float* __restrict__ W, bf16_t* __restrict__ Wt) {
    long long idx = (long long)blockIdx.x * blockDim.x + threadIdx.x;
    if (idx >= (long long)H_DIMC * H_DIMC) return;
    int k = (int)(idx % H_DIMC);
    int n = (int)(idx / H_DIMC);
    Wt[(long long)n * H_DIMC + k] = (bf16_t)W[(long long)k * H_DIMC + n];
}

__global__ void prep_vt_kernel(const bf16_t* __restrict__ V, f16_t* __restrict__ Vt) {
    long long idx = (long long)blockIdx.x * blockDim.x + threadIdx.x;
    if (idx >= (long long)BATCHC * N_HEADSC * HEAD_DIMC * SPADC) return;
    int key = (int)(idx % SPADC);
    int rest = (int)(idx / SPADC);
    int d  = rest % HEAD_DIMC;
    int bh = rest / HEAD_DIMC;
    int h  = bh % N_HEADSC;
    int b  = bh / N_HEADSC;
    f16_t val = (f16_t)0.0f;
    if (key < SEQC)
        val = (f16_t)(float)V[(long long)(b * SEQC + key) * H_DIMC + h * HEAD_DIMC + d];
    Vt[idx] = val;
}

// ---------------- GEMM core: 32x64 tile per wave, double-buffered ----------------
// A operands (2): rows mtile*32 + {0,16} + l16.  B operands (4): cols n4*64 + s2*16.
// Two k-steps per loop iteration keep all buffer indices compile-time constant.
struct GemmAcc { v8f a[8]; };   // acc[s2*2 + mh]

__device__ __forceinline__ void gemm_load_a(const bf16_t* x0, const bf16_t* x1,
                                            int kb, v16bf a[2]) {
    a[0] = ld_pair_bf(x0 + kb, x0 + kb + 16);
    a[1] = ld_pair_bf(x1 + kb, x1 + kb + 16);
}
__device__ __forceinline__ void gemm_load_b(const bf16_t* w0, int kb, v16bf bm[4]) {
    #pragma unroll
    for (int s2 = 0; s2 < 4; ++s2) {
        const bf16_t* wr = w0 + (long long)(s2 * 16) * H_DIMC;
        bm[s2] = ld_pair_bf(wr + kb, wr + kb + 16);
    }
}
__device__ __forceinline__ void gemm_mma(const v16bf a[2], const v16bf bm[4], GemmAcc& acc) {
    #pragma unroll
    for (int s2 = 0; s2 < 4; ++s2)
        #pragma unroll
        for (int mh = 0; mh < 2; ++mh)
            acc.a[s2 * 2 + mh] = __builtin_amdgcn_wmma_f32_16x16x32_bf16(
                false, a[mh], false, bm[s2], (short)0, acc.a[s2 * 2 + mh], false, false);
}

__device__ __forceinline__ void gemm_body(const bf16_t* __restrict__ Xb,
                                          const bf16_t* __restrict__ Wt,
                                          int mtile, int n4, int lane, GemmAcc& acc) {
    int half = lane >> 4;
    int l16  = lane & 15;
    const bf16_t* x0 = Xb + (long long)(mtile * 32 + l16) * H_DIMC;
    const bf16_t* x1 = Xb + (long long)(mtile * 32 + 16 + l16) * H_DIMC;
    const bf16_t* w0 = Wt + (long long)(n4 * 64 + l16) * H_DIMC;

    #pragma unroll
    for (int i = 0; i < 8; ++i)
        #pragma unroll
        for (int r = 0; r < 8; ++r) acc.a[i][r] = 0.0f;

    v16bf aA[2], bA[4], aB[2], bB[4];
    gemm_load_a(x0, x1, half * 8, aA);
    gemm_load_b(w0, half * 8, bA);
    for (int k = 0; k < H_DIMC; k += 64) {
        int kb1 = k + 32 + half * 8;
        gemm_load_a(x0, x1, kb1, aB);            // prefetch k+32
        gemm_load_b(w0, kb1, bB);
        gemm_mma(aA, bA, acc);                   // consume k
        if (k + 64 < H_DIMC) {
            int kb2 = k + 64 + half * 8;
            gemm_load_a(x0, x1, kb2, aA);        // prefetch k+64
            gemm_load_b(w0, kb2, bA);
        }
        gemm_mma(aB, bB, acc);                   // consume k+32
    }
}

// projection GEMM: bf16 output, unguarded (buffers padded, Xb pad rows are zero)
__global__ MMA_BOUNDS void gemm_proj_bf16(const bf16_t* __restrict__ Xb,
                                          const bf16_t* __restrict__ Wt,
                                          const float*  __restrict__ bias,
                                          bf16_t* __restrict__ Out) {
    int lane = threadIdx.x & 31;
    int wid  = blockIdx.x * (blockDim.x >> 5) + (threadIdx.x >> 5);
    if (wid >= MT32 * N4TILES) return;
    int mtile = wid / N4TILES;
    int n4    = wid % N4TILES;
    GemmAcc acc;
    gemm_body(Xb, Wt, mtile, n4, lane, acc);
    int half = lane >> 4, l16 = lane & 15;
    #pragma unroll
    for (int s2 = 0; s2 < 4; ++s2) {
        int n = n4 * 64 + s2 * 16 + l16;
        float bv = bias[n];
        #pragma unroll
        for (int mh = 0; mh < 2; ++mh)
            #pragma unroll
            for (int r = 0; r < 8; ++r) {
                int row = mtile * 32 + mh * 16 + r + half * 8;
                Out[(long long)row * H_DIMC + n] = (bf16_t)(acc.a[s2 * 2 + mh][r] + bv);
            }
    }
}

// output projection: fp32 output into d_out, guarded at 8200 rows
__global__ MMA_BOUNDS void gemm_out_f32(const bf16_t* __restrict__ Xb,
                                        const bf16_t* __restrict__ Wt,
                                        const float*  __restrict__ bias,
                                        float* __restrict__ Out) {
    int lane = threadIdx.x & 31;
    int wid  = blockIdx.x * (blockDim.x >> 5) + (threadIdx.x >> 5);
    if (wid >= MT32 * N4TILES) return;
    int mtile = wid / N4TILES;
    int n4    = wid % N4TILES;
    GemmAcc acc;
    gemm_body(Xb, Wt, mtile, n4, lane, acc);
    int half = lane >> 4, l16 = lane & 15;
    #pragma unroll
    for (int s2 = 0; s2 < 4; ++s2) {
        int n = n4 * 64 + s2 * 16 + l16;
        float bv = bias[n];
        #pragma unroll
        for (int mh = 0; mh < 2; ++mh)
            #pragma unroll
            for (int r = 0; r < 8; ++r) {
                int row = mtile * 32 + mh * 16 + r + half * 8;
                if (row < TOKC)
                    Out[(long long)row * H_DIMC + n] = acc.a[s2 * 2 + mh][r] + bv;
            }
    }
}

// ---------------- 3D rotary on Q and K ----------------
__global__ void rope_kernel(bf16_t* __restrict__ Q, bf16_t* __restrict__ K) {
    long long idx = (long long)blockIdx.x * blockDim.x + threadIdx.x;
    if (idx >= (long long)TOKC * N_HEADSC * 32) return;
    int j   = (int)(idx & 31);
    int h   = (int)((idx >> 5) % N_HEADSC);
    int tok = (int)(idx / (32 * N_HEADSC));
    int pos = tok % SEQC;
    int g   = pos > 0 ? pos - 1 : 0;            // cls_offset = 1 (1025 != 32*32)
    float rowid = (float)(g >> 5);
    float colid = (float)(g & 31);
    float f = __powf(10000.0f, -(float)j * (1.0f / 64.0f));  // 3j/192 = j/64
    float st, ct, sp, cp;
    __sincosf(rowid * f, &st, &ct);
    __sincosf(colid * f, &sp, &cp);

    long long base = (long long)tok * H_DIMC + h * HEAD_DIMC + 3 * j;
    {
        float x0 = (float)Q[base], x1 = (float)Q[base + 1], x2 = (float)Q[base + 2];
        Q[base]     = (bf16_t)( cp * x0 + sp * st * x1 + sp * ct * x2);
        Q[base + 1] = (bf16_t)( ct * x1 - st * x2);
        Q[base + 2] = (bf16_t)(-sp * x0 + cp * st * x1 + cp * ct * x2);
    }
    {
        float x0 = (float)K[base], x1 = (float)K[base + 1], x2 = (float)K[base + 2];
        K[base]     = (bf16_t)( cp * x0 + sp * st * x1 + sp * ct * x2);
        K[base + 1] = (bf16_t)( ct * x1 - st * x2);
        K[base + 2] = (bf16_t)(-sp * x0 + cp * st * x1 + cp * ct * x2);
    }
}

// ---------------- flash attention (transposed): S^T = K Q^T, ctx^T = V^T P^T ----
// One wave handles one (b, h, 32-query tile): K-tile A-operands are shared by the
// two query columns' score WMMAs, V^T A-operands by both PV WMMAs (2x FLOP/byte).
// The C layout of S^T feeds the B operand of the PV WMMA directly (per-lane
// f32->f16 pack, no cross-lane movement). V loads are issued before the softmax
// so they overlap the score WMMAs + exp VALU work.
__global__ MMA_BOUNDS void attn_kernel(const bf16_t* __restrict__ Q,
                                       const bf16_t* __restrict__ K,
                                       const f16_t*  __restrict__ Vt,
                                       bf16_t* __restrict__ Ctx) {
    const float scale = 0.10206207261596575f;   // 96^-0.5
    int lane = threadIdx.x & 31;
    int wid  = blockIdx.x * (blockDim.x >> 5) + (threadIdx.x >> 5);
    if (wid >= BATCHC * N_HEADSC * QT32) return;
    int qt = wid % QT32;
    int h  = (wid / QT32) % N_HEADSC;
    int b  = wid / (QT32 * N_HEADSC);
    int half = lane >> 4;
    int l16  = lane & 15;

    // Q^T as B operands for the two 16-query columns of this wave
    v16bf bq[2][3];
    #pragma unroll
    for (int qi = 0; qi < 2; ++qi) {
        int qtok = b * SEQC + qt * 32 + qi * 16 + l16;
        const bf16_t* qrow = Q + (long long)qtok * H_DIMC + h * HEAD_DIMC;
        #pragma unroll
        for (int c = 0; c < 3; ++c) {
            int kb = c * 32 + half * 8;
            bq[qi][c] = ld_pair_bf(qrow + kb, qrow + kb + 16);
        }
    }

    v8f acc[2][6];
    #pragma unroll
    for (int qi = 0; qi < 2; ++qi)
        #pragma unroll
        for (int dt = 0; dt < 6; ++dt)
            #pragma unroll
            for (int r = 0; r < 8; ++r) acc[qi][dt][r] = 0.0f;
    float m[2] = { -1e30f, -1e30f };
    float l[2] = { 0.0f, 0.0f };

    const f16_t* vbase = Vt + (long long)(b * N_HEADSC + h) * HEAD_DIMC * SPADC;

    for (int kt = 0; kt < KTILES; ++kt) {
        // ---- issue all loads for this key tile up front ----
        v16bf ka[2][3];                           // K rows: tile half t, d-chunk c
        const bf16_t* kr0 = K + (long long)(b * SEQC + kt * 32 + l16) * H_DIMC + h * HEAD_DIMC;
        const bf16_t* kr1 = kr0 + (long long)16 * H_DIMC;
        #pragma unroll
        for (int c = 0; c < 3; ++c) {
            int kb = c * 32 + half * 8;
            ka[0][c] = ld_pair_bf(kr0 + kb, kr0 + kb + 16);
            ka[1][c] = ld_pair_bf(kr1 + kb, kr1 + kb + 16);
        }
        v16h av[6];                               // V^T rows (score-independent)
        #pragma unroll
        for (int dt = 0; dt < 6; ++dt) {
            const f16_t* vrow = vbase + (long long)(dt * 16 + l16) * SPADC + kt * 32 + half * 8;
            av[dt] = ld_pair_h(vrow, vrow + 16);
        }

        // ---- scores: 12 WMMAs (K-tile operands reused across both q columns) ----
        v8f s[2][2];
        #pragma unroll
        for (int qi = 0; qi < 2; ++qi)
            #pragma unroll
            for (int t = 0; t < 2; ++t)
                #pragma unroll
                for (int r = 0; r < 8; ++r) s[qi][t][r] = 0.0f;
        #pragma unroll
        for (int c = 0; c < 3; ++c)
            #pragma unroll
            for (int qi = 0; qi < 2; ++qi) {
                s[qi][0] = __builtin_amdgcn_wmma_f32_16x16x32_bf16(
                    false, ka[0][c], false, bq[qi][c], (short)0, s[qi][0], false, false);
                s[qi][1] = __builtin_amdgcn_wmma_f32_16x16x32_bf16(
                    false, ka[1][c], false, bq[qi][c], (short)0, s[qi][1], false, false);
            }

        // ---- online softmax per query column ----
        int key0 = kt * 32 + half * 8;
        v16h pb[2];
        #pragma unroll
        for (int qi = 0; qi < 2; ++qi) {
            float tm = -1e30f;
            #pragma unroll
            for (int r = 0; r < 8; ++r) {
                float x0 = s[qi][0][r] * scale; if (key0 + r      >= SEQC) x0 = -1e30f;
                float x1 = s[qi][1][r] * scale; if (key0 + 16 + r >= SEQC) x1 = -1e30f;
                s[qi][0][r] = x0; s[qi][1][r] = x1;
                tm = fmaxf(tm, fmaxf(x0, x1));
            }
            tm = fmaxf(tm, __shfl_xor(tm, 16, 32));   // other 16 keys at lane^16
            float mn    = fmaxf(m[qi], tm);
            float alpha = __expf(m[qi] - mn);
            float rs = 0.0f;
            #pragma unroll
            for (int r = 0; r < 8; ++r) {
                float p0 = __expf(s[qi][0][r] - mn);
                float p1 = __expf(s[qi][1][r] - mn);
                rs += p0 + p1;
                pb[qi][r]     = (f16_t)p0;            // B elem i<8  -> key half*8+i
                pb[qi][r + 8] = (f16_t)p1;            // B elem i>=8 -> key half*8+(i-8)+16
            }
            rs += __shfl_xor(rs, 16, 32);
            l[qi] = l[qi] * alpha + rs;
            m[qi] = mn;
            #pragma unroll
            for (int dt = 0; dt < 6; ++dt)
                #pragma unroll
                for (int r = 0; r < 8; ++r) acc[qi][dt][r] *= alpha;
        }

        // ---- ctx^T += V^T P^T : 12 WMMAs (V operands reused across q columns) ----
        #pragma unroll
        for (int dt = 0; dt < 6; ++dt)
            #pragma unroll
            for (int qi = 0; qi < 2; ++qi)
                acc[qi][dt] = __builtin_amdgcn_wmma_f32_16x16x32_f16(
                    false, av[dt], false, pb[qi], (short)0, acc[qi][dt], false, false);
    }

    #pragma unroll
    for (int qi = 0; qi < 2; ++qi) {
        float inv = 1.0f / l[qi];
        int qpos = qt * 32 + qi * 16 + l16;
        if (qpos < SEQC) {
            long long orow = (long long)(b * SEQC + qpos) * H_DIMC + h * HEAD_DIMC;
            #pragma unroll
            for (int dt = 0; dt < 6; ++dt)
                #pragma unroll
                for (int r = 0; r < 8; ++r)
                    Ctx[orow + dt * 16 + r + half * 8] = (bf16_t)(acc[qi][dt][r] * inv);
        }
    }
}

// ---------------- host launcher ----------------
extern "C" void kernel_launch(void* const* d_in, const int* in_sizes, int n_in,
                              void* d_out, int out_size, void* d_ws, size_t ws_size,
                              hipStream_t stream) {
    (void)in_sizes; (void)n_in; (void)out_size; (void)ws_size;
    const float* hs = (const float*)d_in[0];
    const float* Wq = (const float*)d_in[1];
    const float* bq = (const float*)d_in[2];
    const float* Wk = (const float*)d_in[3];
    const float* bk = (const float*)d_in[4];
    const float* Wv = (const float*)d_in[5];
    const float* bv = (const float*)d_in[6];
    const float* Wo = (const float*)d_in[7];
    const float* bo = (const float*)d_in[8];
    float* out = (float*)d_out;

    char* ws = (char*)d_ws;
    const long long SZ_XB = (long long)MPADC * H_DIMC * 2;                         // 19,464,192
    const long long SZ_W  = (long long)H_DIMC * H_DIMC * 2;                        //  2,654,208
    const long long SZ_VT = (long long)BATCHC * N_HEADSC * HEAD_DIMC * SPADC * 2;  // 19,464,192
    bf16_t* Xb  = (bf16_t*)(ws);
    bf16_t* Wqt = (bf16_t*)(ws + SZ_XB);
    bf16_t* Wkt = (bf16_t*)(ws + SZ_XB + SZ_W);
    bf16_t* Wvt = (bf16_t*)(ws + SZ_XB + 2 * SZ_W);
    bf16_t* Wot = (bf16_t*)(ws + SZ_XB + 3 * SZ_W);
    bf16_t* Qb  = (bf16_t*)(ws + SZ_XB + 4 * SZ_W);
    bf16_t* Kb  = (bf16_t*)(ws + 2 * SZ_XB + 4 * SZ_W);
    bf16_t* Vb  = (bf16_t*)(ws + 3 * SZ_XB + 4 * SZ_W);
    f16_t*  Vtp = (f16_t*) (ws + 4 * SZ_XB + 4 * SZ_W);
    bf16_t* Ctx = (bf16_t*)(ws + 4 * SZ_XB + 4 * SZ_W + SZ_VT);

    const int TB  = 256;   // simple elementwise kernels
    const int TBW = 128;   // WMMA kernels: 4 waves/block, 1 wave per SIMD32
    // 1) convert/pad X, transpose/convert weights
    {
        long long n = (long long)MPADC * H_DIMC;
        prep_x_kernel<<<(unsigned)((n + TB - 1) / TB), TB, 0, stream>>>(hs, Xb);
        long long nw = (long long)H_DIMC * H_DIMC;
        unsigned gw = (unsigned)((nw + TB - 1) / TB);
        prep_w_kernel<<<gw, TB, 0, stream>>>(Wq, Wqt);
        prep_w_kernel<<<gw, TB, 0, stream>>>(Wk, Wkt);
        prep_w_kernel<<<gw, TB, 0, stream>>>(Wv, Wvt);
        prep_w_kernel<<<gw, TB, 0, stream>>>(Wo, Wot);
    }
    // 2) QKV projections (WMMA bf16, 32x64 tiles, double-buffered)
    {
        int tasks = MT32 * N4TILES;              // 4626 wave-tasks
        unsigned blocks = (unsigned)((tasks + (TBW / 32) - 1) / (TBW / 32));
        gemm_proj_bf16<<<blocks, TBW, 0, stream>>>(Xb, Wqt, bq, Qb);
        gemm_proj_bf16<<<blocks, TBW, 0, stream>>>(Xb, Wkt, bk, Kb);
        gemm_proj_bf16<<<blocks, TBW, 0, stream>>>(Xb, Wvt, bv, Vb);
    }
    // 3) rotary on Q,K
    {
        long long n = (long long)TOKC * N_HEADSC * 32;
        rope_kernel<<<(unsigned)((n + TB - 1) / TB), TB, 0, stream>>>(Qb, Kb);
    }
    // 4) V -> V^T (f16, zero-padded keys)
    {
        long long n = (long long)BATCHC * N_HEADSC * HEAD_DIMC * SPADC;
        prep_vt_kernel<<<(unsigned)((n + TB - 1) / TB), TB, 0, stream>>>(Vb, Vtp);
    }
    // 5) flash attention (32 queries/wave; WMMA bf16 scores + f16 PV)
    {
        int tasks = BATCHC * N_HEADSC * QT32;    // 3168 wave-tasks
        unsigned blocks = (unsigned)((tasks + (TBW / 32) - 1) / (TBW / 32));
        attn_kernel<<<blocks, TBW, 0, stream>>>(Qb, Kb, Vtp, Ctx);
    }
    // 6) output projection -> fp32 d_out (guarded at 8200 rows)
    {
        int tasks = MT32 * N4TILES;
        unsigned blocks = (unsigned)((tasks + (TBW / 32) - 1) / (TBW / 32));
        gemm_out_f32<<<blocks, TBW, 0, stream>>>(Ctx, Wot, bo, out);
    }
}
